// LayerRecurrent_16655883174401
// MI455X (gfx1250) — matrix operand order repaired and checked by reading, not compiled, not run on hardware
//
#include <hip/hip_runtime.h>
#include <hip/hip_bf16.h>
#include <math.h>

// ---------------------------------------------------------------------------
// Echo-state recurrence for MI455X (gfx1250, wave32, WMMA f32_16x16x32_f16)
//   DIM=1024, INPUT_DIM=300 (padded 320), T=512, B=64, LEAK=0.5
// ---------------------------------------------------------------------------

#define DIM   1024
#define INK   300
#define INKP  320
#define TLEN  512
#define BSZ   64

typedef __attribute__((ext_vector_type(16))) _Float16 v16h;
typedef __attribute__((ext_vector_type(8)))  float    v8f;

// ---------------------------------------------------------------------------
// One K=32 chunk of a 16x16 WMMA tile.
//  A element (m,k) at Apanel[m*lda + k]   (m = batch row within tile)
//  B element (k,n) at Bpanel[n*ldb + k]   (n = output dim within tile; B is
//                                          W stored row-major, so column n of
//                                          the K x N operand is W row n)
// VGPR packing per CDNA5 ISA 7.12.2 (16-bit A 16x32; B 32x16 by the sparse-B
// analogy: lanes 0-15 hold K=0-7,16-23, lanes 16-31 hold K=8-15,24-31, n = lane & 15).
// ---------------------------------------------------------------------------
__device__ __forceinline__ v8f mm_chunk(const _Float16* __restrict__ Apanel, int lda,
                                        const _Float16* __restrict__ Bpanel, int ldb,
                                        int kb, v8f c, int lane)
{
    const int half = lane >> 4;   // 0: lanes 0-15, 1: lanes 16-31
    const int mr   = lane & 15;   // A row m  /  B col n
    union { v16h v; unsigned u[8]; } ua, ub;

    const _Float16* ap = Apanel + mr * lda + kb;
    #pragma unroll
    for (int v = 0; v < 4; ++v)       // VGPR 0-3: K = half*8 + {0..7}
        ua.u[v] = *(const unsigned*)(ap + half * 8 + 2 * v);
    #pragma unroll
    for (int v = 0; v < 4; ++v)       // VGPR 4-7: K = 16 + half*8 + {0..7}
        ua.u[4 + v] = *(const unsigned*)(ap + 16 + half * 8 + 2 * v);

    const _Float16* bp = Bpanel + mr * ldb + kb + half * 8;
    #pragma unroll
    for (int v = 0; v < 8; ++v)       // K = half*8 + {0..7}, then 16 + half*8 + {0..7}
        ub.u[v] = *(const unsigned*)(bp + (v >> 2) * 16 + 2 * (v & 3));

    // (neg_a, A, neg_b, B, c_mod, C, reuse_a, reuse_b)
    return __builtin_amdgcn_wmma_f32_16x16x32_f16(false, ua.v, false, ub.v,
                                                  (short)0, c, false, false);
}

// ---------------------------------------------------------------------------
// One recurrence timestep.
//   x_new = mask * (0.5*tanh(emb_t @ Uhᵀ + bias + x @ Whᵀ) + 0.5*x)
// grid = 64 N-tiles (16 dims each), block = 128 threads = 4 waves,
// wave w owns batch-tile rows [16w, 16w+16).
// ---------------------------------------------------------------------------
__global__ __launch_bounds__(128) void k_esn_step(
    const _Float16* __restrict__ Wh,      // (DIM, DIM) f16, row-major (d, k)
    const _Float16* __restrict__ Uh,      // (DIM, INKP) f16, zero-padded K
    const _Float16* __restrict__ Eh,      // (T, B, INKP) f16, zero-padded K
    const float*    __restrict__ bias,    // (DIM)
    const int*      __restrict__ lengths, // (B)
    const float*    __restrict__ Xprev,   // (B, DIM) f32 state (read)
    const _Float16* __restrict__ XhPrev,  // (B, DIM) f16 state (read)
    float*          __restrict__ Xnext,   // (B, DIM) f32 state (write)
    _Float16*       __restrict__ XhNext,  // (B, DIM) f16 state (write)
    float*          __restrict__ states,  // (B, T, DIM) output
    int t)
{
    const int lane  = threadIdx.x & 31;
    const int wave  = threadIdx.x >> 5;   // batch tile 0..3
    const int tileN = blockIdx.x;         // dim tile 0..63

    // hint: pull this tile's W panel toward the WGP early (L2-resident data)
    __builtin_prefetch(Wh + (size_t)(tileN * 16) * DIM, 0, 0);

    // ---- recurrent part: acc += x @ Wᵀ, K = 1024 ----
    const _Float16* Apanel = XhPrev + (size_t)(wave * 16) * DIM;
    const _Float16* Bpanel = Wh + (size_t)(tileN * 16) * DIM;
    v8f acc = {};
    #pragma unroll 4
    for (int kb = 0; kb < DIM; kb += 32)
        acc = mm_chunk(Apanel, DIM, Bpanel, DIM, kb, acc, lane);

    // ---- fused input projection: acc += emb_t @ input_wᵀ, K = 320 (padded) ----
    const _Float16* Ae = Eh + (size_t)t * (BSZ * INKP) + (size_t)(wave * 16) * INKP;
    const _Float16* Be = Uh + (size_t)(tileN * 16) * INKP;
    #pragma unroll
    for (int kb = 0; kb < INKP; kb += 32)
        acc = mm_chunk(Ae, INKP, Be, INKP, kb, acc, lane);

    // ---- epilogue: bias, tanh, leaky integration, length mask ----
    const int n = lane & 15;
    const int d = tileN * 16 + n;
    const float bd = bias[d];
    #pragma unroll
    for (int v = 0; v < 8; ++v) {
        const int m = v + 8 * (lane >> 4);    // C/D layout: VGPR v -> row m
        const int b = wave * 16 + m;
        const size_t sidx = (size_t)b * DIM + d;
        const float xp  = Xprev[sidx];
        const float pre = acc[v] + bd;
        float xn = 0.5f * tanhf(pre) + 0.5f * xp;
        if (lengths[b] <= t) xn = 0.0f;
        Xnext[sidx]  = xn;
        XhNext[sidx] = (_Float16)xn;
        states[((size_t)b * TLEN + t) * DIM + d] = xn;
    }
}

// ---------------------------------------------------------------------------
// Prologue kernels
// ---------------------------------------------------------------------------
__global__ void k_zero_len(int* __restrict__ len)
{
    if (threadIdx.x < BSZ) len[threadIdx.x] = 0;
}

// one wave per (t,b): lengths[b] += (sum_i emb[t,b,i] != 0)
__global__ __launch_bounds__(32) void k_lengths(const float* __restrict__ emb,
                                                int* __restrict__ len)
{
    const int tb = blockIdx.x;            // t*B + b
    const int b  = tb & (BSZ - 1);
    const float* row = emb + (size_t)tb * INK;
    float s = 0.0f;
    for (int i = threadIdx.x; i < INK; i += 32) s += row[i];
    #pragma unroll
    for (int off = 16; off > 0; off >>= 1) s += __shfl_down(s, off, 32);
    if (threadIdx.x == 0 && s != 0.0f) atomicAdd(&len[b], 1);
}

__global__ void k_f32_to_f16(const float* __restrict__ src,
                             _Float16* __restrict__ dst, int n)
{
    int i = blockIdx.x * blockDim.x + threadIdx.x;
    if (i < n) dst[i] = (_Float16)src[i];
}

// convert + pad inner K dim: (rows, in_k) f32 -> (rows, out_k) f16, zeros past in_k
__global__ void k_pad_convert(const float* __restrict__ src,
                              _Float16* __restrict__ dst,
                              int rows, int in_k, int out_k)
{
    int i = blockIdx.x * blockDim.x + threadIdx.x;
    if (i >= rows * out_k) return;
    int r = i / out_k, k = i - r * out_k;
    dst[i] = (k < in_k) ? (_Float16)src[(size_t)r * in_k + k] : (_Float16)0.0f;
}

__global__ void k_init_state(const float* __restrict__ init,
                             float* __restrict__ Xf, _Float16* __restrict__ Xh)
{
    int i = blockIdx.x * blockDim.x + threadIdx.x;
    if (i >= BSZ * DIM) return;
    float v = init[i & (DIM - 1)];
    Xf[i] = v;
    Xh[i] = (_Float16)v;
}

__global__ void k_copy_len(const int* __restrict__ len, int* __restrict__ out)
{
    if (threadIdx.x < BSZ) out[threadIdx.x] = len[threadIdx.x];
}

// ---------------------------------------------------------------------------
// Host launch
// ---------------------------------------------------------------------------
extern "C" void kernel_launch(void* const* d_in, const int* in_sizes, int n_in,
                              void* d_out, int out_size, void* d_ws, size_t ws_size,
                              hipStream_t stream)
{
    (void)in_sizes; (void)n_in; (void)out_size; (void)ws_size;

    const float* emb     = (const float*)d_in[0];  // (T, B, 300)
    const float* input_w = (const float*)d_in[1];  // (DIM, 300)
    const float* layer_w = (const float*)d_in[2];  // (DIM, DIM)
    const float* bias    = (const float*)d_in[3];  // (DIM)
    const float* init    = (const float*)d_in[4];  // (DIM)

    float* states = (float*)d_out;                              // (B, T, DIM)
    int*   len_out = (int*)((float*)d_out + (size_t)BSZ * TLEN * DIM);

    // workspace layout (all sizes multiples of 256 B); total ~24.6 MB
    char* ws = (char*)d_ws;
    const size_t WH_B  = (size_t)DIM * DIM * sizeof(_Float16);        // 2 MB
    const size_t UH_B  = (size_t)DIM * INKP * sizeof(_Float16);       // 640 KB
    const size_t EH_B  = (size_t)TLEN * BSZ * INKP * sizeof(_Float16);// 20 MB
    const size_t XF_B  = (size_t)BSZ * DIM * sizeof(float);           // 256 KB
    const size_t XH_B  = (size_t)BSZ * DIM * sizeof(_Float16);        // 128 KB

    _Float16* Wh  = (_Float16*)(ws);
    _Float16* Uh  = (_Float16*)(ws + WH_B);
    _Float16* Eh  = (_Float16*)(ws + WH_B + UH_B);
    float*    Xf0 = (float*)   (ws + WH_B + UH_B + EH_B);
    float*    Xf1 = (float*)   (ws + WH_B + UH_B + EH_B + XF_B);
    _Float16* Xh0 = (_Float16*)(ws + WH_B + UH_B + EH_B + 2 * XF_B);
    _Float16* Xh1 = (_Float16*)(ws + WH_B + UH_B + EH_B + 2 * XF_B + XH_B);
    int*      len = (int*)     (ws + WH_B + UH_B + EH_B + 2 * XF_B + 2 * XH_B);

    // --- prologue ---
    k_zero_len<<<1, 64, 0, stream>>>(len);
    k_lengths<<<TLEN * BSZ, 32, 0, stream>>>(emb, len);
    k_f32_to_f16<<<(DIM * DIM + 255) / 256, 256, 0, stream>>>(layer_w, Wh, DIM * DIM);
    k_pad_convert<<<(DIM * INKP + 255) / 256, 256, 0, stream>>>(input_w, Uh, DIM, INK, INKP);
    k_pad_convert<<<(TLEN * BSZ * INKP + 255) / 256, 256, 0, stream>>>(emb, Eh, TLEN * BSZ, INK, INKP);
    k_init_state<<<(BSZ * DIM + 255) / 256, 256, 0, stream>>>(init, Xf0, Xh0);

    // --- 512 sequential steps (ping-pong state buffers) ---
    for (int t = 0; t < TLEN; ++t) {
        const float*    xpf = (t & 1) ? Xf1 : Xf0;
        const _Float16* xph = (t & 1) ? Xh1 : Xh0;
        float*          xnf = (t & 1) ? Xf0 : Xf1;
        _Float16*       xnh = (t & 1) ? Xh0 : Xh1;
        k_esn_step<<<DIM / 16, 128, 0, stream>>>(Wh, Uh, Eh, bias, len,
                                                 xpf, xph, xnf, xnh, states, t);
    }

    // --- second tuple output: lengths (bit-copied into the f32 tail) ---
    k_copy_len<<<1, 64, 0, stream>>>(len, len_out);
}
